// OndiAttention_54692113547865
// MI455X (gfx1250) — matrix operand order repaired and checked by reading, not compiled, not run on hardware
//
#include <hip/hip_runtime.h>

// ---------------------------------------------------------------------------
// MHA forward for MI455X (gfx1250, wave32, WMMA).
// B=2, S=2048, D=2048, H=16, hd=128. fp32 in/out, bf16 WMMA compute, f32 acc.
// GEMMs + attention: LDS double-buffered with async global->LDS staging.
// ---------------------------------------------------------------------------

#define S_LEN  2048
#define D_DIM  2048
#define K_DIM  2048
#define NHEAD  16
#define HDIM   128
#define BSZ    2
#define MROWS  (BSZ * S_LEN)   // 4096

#define USE_ASYNC 1            // gfx1250 GLOBAL_LOAD_ASYNC_TO_LDS_B128 path

typedef __attribute__((ext_vector_type(16))) __bf16       v16bf;
typedef __attribute__((ext_vector_type(8)))  float        v8f;
typedef __attribute__((ext_vector_type(4))) unsigned int  u32x4;

// f32 -> bf16 round-to-nearest-even
__device__ __forceinline__ unsigned short f2bf(float f) {
    unsigned int u = __float_as_uint(f);
    u += 0x7FFFu + ((u >> 16) & 1u);
    return (unsigned short)(u >> 16);
}

// Build a 16-element bf16 fragment from two 16-byte-aligned 8-element groups.
__device__ __forceinline__ v16bf load_frag(const unsigned short* p0,
                                           const unsigned short* p1) {
    union { v16bf v; u32x4 q[2]; } u;
    u.q[0] = *reinterpret_cast<const u32x4*>(p0);
    u.q[1] = *reinterpret_cast<const u32x4*>(p1);
    return u.v;
}

__device__ __forceinline__ v8f wmma_bf16(v16bf a, v16bf b, v8f c) {
    return __builtin_amdgcn_wmma_f32_16x16x32_bf16(
        /*neg_a=*/false, a, /*neg_b=*/false, b,
        /*c_mod=*/(short)0, c, /*reuse_a=*/false, /*reuse_b=*/false);
}

// 16B global -> LDS copy. Async path: GLOBAL_LOAD_ASYNC_TO_LDS_B128
// (ISA 15.18.3 op 98; VDST = LDS byte offset = low 32 bits of generic pointer
// per ISA 10.2 aperture rules), tracked by ASYNCcnt.
__device__ __forceinline__ void cp16(void* ldst, const void* gsrc) {
#if USE_ASYNC
    asm volatile("global_load_async_to_lds_b128 %0, %1, off"
                 :: "v"((unsigned)(unsigned long long)ldst), "v"(gsrc)
                 : "memory");
#else
    *(u32x4*)ldst = *(const u32x4*)gsrc;
#endif
}

__device__ __forceinline__ void wait_async0() {
#if USE_ASYNC
    asm volatile("s_wait_asynccnt 0x0" ::: "memory");
#endif
}

// ---------------------------------------------------------------------------
__global__ __launch_bounds__(256) void cvt_f32_bf16(const float* __restrict__ in,
                                                    unsigned short* __restrict__ out,
                                                    int n) {
    int i = blockIdx.x * blockDim.x + threadIdx.x;
    int stride = gridDim.x * blockDim.x;
    for (; i < n; i += stride) out[i] = f2bf(in[i]);
}

// ---------------------------------------------------------------------------
// C = A (MROWS x 2048 bf16, row-major) * W^T, W (2048 x 2048 bf16 row-major).
// Block: 8 waves -> 128x128 tile. Wave (wm,wn) -> 64x32 (4x2 WMMA subtiles).
// LDS: ping-pong 128x32 A-panel + 128x32 B-panel staged with async copies.
// mode 0: bf16 row-major | mode 1: bf16 Vt[b][h][hd][S] | mode 2: f32 row-major
__global__ __launch_bounds__(256) void gemm_bf16_wmma(
    const unsigned short* __restrict__ A,
    const unsigned short* __restrict__ W,
    unsigned short* __restrict__ outB,
    float* __restrict__ outF,
    int mode)
{
    __shared__ __align__(16) unsigned short Abuf[2][128][32];
    __shared__ __align__(16) unsigned short Bbuf[2][128][32];

    const int t    = threadIdx.x;
    const int lane = t & 31, half = lane >> 4, lm = lane & 15;
    const int w    = t >> 5;
    const int wm   = w >> 2;            // 0..1 -> 64-row slab
    const int wn   = w & 3;             // 0..3 -> 32-col slab

    const int bm = (int)blockIdx.x >> 4;    // 32 M-blocks of 128
    const int bn = (int)blockIdx.x & 15;    // 16 N-blocks of 128

    const unsigned short* Ag = A + (size_t)(bm * 128) * K_DIM;
    const unsigned short* Wg = W + (size_t)(bn * 128) * K_DIM;

    v8f acc[4][2];
    #pragma unroll
    for (int mi = 0; mi < 4; ++mi)
        #pragma unroll
        for (int ni = 0; ni < 2; ++ni)
            #pragma unroll
            for (int r = 0; r < 8; ++r) acc[mi][ni][r] = 0.0f;

    // Panel staging: 512 16-byte chunks per panel, thread covers chunks t, t+256.
    auto stage = [&](int buf, int k0) {
        #pragma unroll
        for (int c0 = 0; c0 < 2; ++c0) {
            int c   = t + c0 * 256;
            int row = c >> 2;
            int col = (c & 3) * 8;
            cp16(&Abuf[buf][row][col], Ag + (size_t)row * K_DIM + k0 + col);
            cp16(&Bbuf[buf][row][col], Wg + (size_t)row * K_DIM + k0 + col);
        }
    };

    stage(0, 0);
    wait_async0();
    __syncthreads();

    const int KSTEPS = K_DIM / 32;
    for (int ks = 0; ks < KSTEPS; ++ks) {
        int cur = ks & 1;
        if (ks + 1 < KSTEPS) stage(1 - cur, (ks + 1) * 32);  // overlap with WMMA

        v16bf af[4], bfr[2];
        #pragma unroll
        for (int mi = 0; mi < 4; ++mi) {
            const unsigned short* p = &Abuf[cur][wm * 64 + mi * 16 + lm][half * 8];
            af[mi] = load_frag(p, p + 16);
        }
        #pragma unroll
        for (int ni = 0; ni < 2; ++ni) {
            const unsigned short* p = &Bbuf[cur][wn * 32 + ni * 16 + lm][half * 8];
            bfr[ni] = load_frag(p, p + 16);
        }
        #pragma unroll
        for (int mi = 0; mi < 4; ++mi)
            #pragma unroll
            for (int ni = 0; ni < 2; ++ni)
                acc[mi][ni] = wmma_bf16(af[mi], bfr[ni], acc[mi][ni]);

        if (ks + 1 < KSTEPS) {
            wait_async0();
            __syncthreads();
        }
    }

    #pragma unroll
    for (int mi = 0; mi < 4; ++mi) {
        #pragma unroll
        for (int ni = 0; ni < 2; ++ni) {
            #pragma unroll
            for (int r = 0; r < 8; ++r) {
                int gr = bm * 128 + wm * 64 + mi * 16 + r + half * 8;
                int gc = bn * 128 + wn * 32 + ni * 16 + lm;
                float v = acc[mi][ni][r];
                if (mode == 2) {
                    outF[(size_t)gr * D_DIM + gc] = v;
                } else if (mode == 0) {
                    outB[(size_t)gr * D_DIM + gc] = f2bf(v);
                } else { // Vt[b][h][d][s]
                    int h = gc >> 7, d = gc & 127;
                    int s = gr & (S_LEN - 1), b = gr >> 11;
                    outB[(((size_t)(b * NHEAD + h)) * HDIM + d) * S_LEN + s] = f2bf(v);
                }
            }
        }
    }
}

// ---------------------------------------------------------------------------
// Flash attention, block-cooperative: all 8 waves of a block share one (b,h)
// and own 8 consecutive 16-query tiles (a 128-query slab). K/V panels for each
// 32-key pair are staged into double-buffered LDS with async copies and shared
// by all waves. Online softmax, causal mask, P transposed via per-wave slab.
__global__ __launch_bounds__(256) void attn_wmma(
    const unsigned short* __restrict__ Qb,   // [B*S][D] bf16
    const unsigned short* __restrict__ Kb,   // [B*S][D] bf16
    const unsigned short* __restrict__ Vt,   // [B][H][hd][S] bf16
    unsigned short* __restrict__ ctx)        // [B*S][D] bf16
{
    __shared__ __align__(16) unsigned short Klds[2][32][128];  // [key][dim]
    __shared__ __align__(16) unsigned short Vlds[2][128][32];  // [dim][key]
    __shared__ __align__(16) unsigned short plds[8][16][32];   // per-wave P slab

    const int t    = threadIdx.x;
    const int w    = t >> 5;
    const int lane = t & 31, half = lane >> 4, lm = lane & 15;

    const int blk = (int)blockIdx.x;     // 512 blocks
    const int sb  = blk & 15;            // 16 slabs of 128 queries per (b,h)
    const int h   = (blk >> 4) & (NHEAD - 1);
    const int b   = blk >> 8;
    const int q0b = sb * 128;            // slab base
    const int q0  = q0b + w * 16;        // this wave's query tile

    const float rscale  = 0.08838834764831845f;   // 1/sqrt(128)
    const float NEG_INF = -__builtin_inff();

    const unsigned short* kbase = Kb + (size_t)(b * S_LEN) * D_DIM + h * HDIM;
    const unsigned short* vbase = Vt + ((size_t)(b * NHEAD + h)) * HDIM * S_LEN;

    // Stage 32-key K/V panels (8KB each) for key-pair kp into buffer buf.
    auto stage = [&](int buf, int kp) {
        int k0 = kp * 32;
        #pragma unroll
        for (int c0 = 0; c0 < 2; ++c0) {
            int c = t + c0 * 256;                 // 0..511 chunks of 16B
            int kr = c >> 4, kc = (c & 15) * 8;   // K: 32 rows x 128 elems
            cp16(&Klds[buf][kr][kc], kbase + (size_t)(k0 + kr) * D_DIM + kc);
            int vr = c >> 2, vc = (c & 3) * 8;    // V: 128 rows x 32 elems
            cp16(&Vlds[buf][vr][vc], vbase + (size_t)vr * S_LEN + k0 + vc);
        }
    };

    // Q fragments (16 queries x 128 dims) live in registers for the whole loop
    v16bf qf[4];
    const unsigned short* qrow =
        Qb + (size_t)(b * S_LEN + q0 + lm) * D_DIM + h * HDIM;
    #pragma unroll
    for (int kk = 0; kk < 4; ++kk)
        qf[kk] = load_frag(qrow + kk * 32 + half * 8,
                           qrow + kk * 32 + half * 8 + 16);

    v8f o[8];
    #pragma unroll
    for (int nt = 0; nt < 8; ++nt)
        #pragma unroll
        for (int r = 0; r < 8; ++r) o[nt][r] = 0.0f;

    float mi[8], li[8], alpha[8];
    #pragma unroll
    for (int r = 0; r < 8; ++r) { mi[r] = NEG_INF; li[r] = 0.0f; }

    const int kpmax = (q0b >> 5) + 4;    // (q0b + 128) / 32 key-pairs

    stage(0, 0);
    wait_async0();
    __syncthreads();

    for (int kp = 0; kp < kpmax; ++kp) {
        int cur = kp & 1;
        if (kp + 1 < kpmax) stage(1 - cur, kp + 1);   // overlap with WMMA

        float sv[2][8];
        #pragma unroll
        for (int tt = 0; tt < 2; ++tt) {
            int kt = kp * 32 + tt * 16;
            v8f sc;
            #pragma unroll
            for (int r = 0; r < 8; ++r) sc[r] = 0.0f;
            #pragma unroll
            for (int kk = 0; kk < 4; ++kk) {
                const unsigned short* kr =
                    &Klds[cur][tt * 16 + lm][kk * 32 + half * 8];
                v16bf kf = load_frag(kr, kr + 16);
                sc = wmma_bf16(qf[kk], kf, sc);
            }
            #pragma unroll
            for (int r = 0; r < 8; ++r) {
                float s = sc[r] * rscale;
                if (kt + lm > q0 + r + half * 8) s = NEG_INF;   // causal
                sv[tt][r] = s;
            }
        }

        // online softmax: row-wise reductions across the 16-lane half-groups
        #pragma unroll
        for (int r = 0; r < 8; ++r) {
            float rm = fmaxf(sv[0][r], sv[1][r]);
            rm = fmaxf(rm, __shfl_xor(rm, 1, 32));
            rm = fmaxf(rm, __shfl_xor(rm, 2, 32));
            rm = fmaxf(rm, __shfl_xor(rm, 4, 32));
            rm = fmaxf(rm, __shfl_xor(rm, 8, 32));
            float mn = fmaxf(mi[r], rm);
            float al = __expf(mi[r] - mn);
            float p0 = __expf(sv[0][r] - mn);
            float p1 = __expf(sv[1][r] - mn);
            float rs = p0 + p1;
            rs += __shfl_xor(rs, 1, 32);
            rs += __shfl_xor(rs, 2, 32);
            rs += __shfl_xor(rs, 4, 32);
            rs += __shfl_xor(rs, 8, 32);
            li[r] = li[r] * al + rs;
            mi[r] = mn;
            alpha[r] = al;
            sv[0][r] = p0;
            sv[1][r] = p1;
        }

        #pragma unroll
        for (int nt = 0; nt < 8; ++nt)
            #pragma unroll
            for (int r = 0; r < 8; ++r) o[nt][r] *= alpha[r];

        // transpose P (C/D layout -> A layout) through per-wave LDS slab
        #pragma unroll
        for (int r = 0; r < 8; ++r) {
            plds[w][r + half * 8][lm]      = f2bf(sv[0][r]);
            plds[w][r + half * 8][lm + 16] = f2bf(sv[1][r]);
        }
        asm volatile("s_wait_dscnt 0x0" ::: "memory");
        v16bf pf = load_frag(&plds[w][lm][half * 8],
                             &plds[w][lm][half * 8 + 16]);

        // O += P * V from the shared V panel
        #pragma unroll
        for (int nt = 0; nt < 8; ++nt) {
            const unsigned short* vr = &Vlds[cur][nt * 16 + lm][half * 8];
            v16bf vf = load_frag(vr, vr + 16);
            o[nt] = wmma_bf16(pf, vf, o[nt]);
        }

        if (kp + 1 < kpmax) {
            wait_async0();
            __syncthreads();
        }
    }

    #pragma unroll
    for (int r = 0; r < 8; ++r) {
        float inv = 1.0f / li[r];
        size_t rowbase = (size_t)(b * S_LEN + q0 + r + half * 8) * D_DIM + h * HDIM;
        #pragma unroll
        for (int nt = 0; nt < 8; ++nt)
            ctx[rowbase + nt * 16 + lm] = f2bf(o[nt][r] * inv);
    }
}

// ---------------------------------------------------------------------------
extern "C" void kernel_launch(void* const* d_in, const int* in_sizes, int n_in,
                              void* d_out, int out_size, void* d_ws, size_t ws_size,
                              hipStream_t stream) {
    (void)in_sizes; (void)n_in; (void)out_size; (void)ws_size;
    const float* x  = (const float*)d_in[0];
    const float* Wq = (const float*)d_in[1];
    const float* Wk = (const float*)d_in[2];
    const float* Wv = (const float*)d_in[3];
    const float* Wo = (const float*)d_in[4];

    const size_t SZ_X = (size_t)MROWS * D_DIM;
    const size_t SZ_W = (size_t)D_DIM * D_DIM;

    unsigned short* xb  = (unsigned short*)d_ws;
    unsigned short* wqb = xb  + SZ_X;
    unsigned short* wkb = wqb + SZ_W;
    unsigned short* wvb = wkb + SZ_W;
    unsigned short* wob = wvb + SZ_W;
    unsigned short* Qb  = wob + SZ_W;
    unsigned short* Kb  = Qb  + SZ_X;
    unsigned short* Vt  = Kb  + SZ_X;
    unsigned short* ctx = Vt  + SZ_X;
    // total workspace: 117,440,512 bytes

    cvt_f32_bf16<<<2048, 256, 0, stream>>>(x,  xb,  (int)SZ_X);
    cvt_f32_bf16<<<1024, 256, 0, stream>>>(Wq, wqb, (int)SZ_W);
    cvt_f32_bf16<<<1024, 256, 0, stream>>>(Wk, wkb, (int)SZ_W);
    cvt_f32_bf16<<<1024, 256, 0, stream>>>(Wv, wvb, (int)SZ_W);
    cvt_f32_bf16<<<1024, 256, 0, stream>>>(Wo, wob, (int)SZ_W);

    // (M/128)*(N/128) = 32*16 = 512 blocks, 8 waves each
    gemm_bf16_wmma<<<512, 256, 0, stream>>>(xb, wqb, Qb, nullptr, 0);
    gemm_bf16_wmma<<<512, 256, 0, stream>>>(xb, wkb, Kb, nullptr, 0);
    gemm_bf16_wmma<<<512, 256, 0, stream>>>(xb, wvb, Vt, nullptr, 1);

    // B * H * 16 slabs = 512 blocks; 8 waves x 16 queries each
    attn_wmma<<<512, 256, 0, stream>>>(Qb, Kb, Vt, ctx);

    gemm_bf16_wmma<<<512, 256, 0, stream>>>(ctx, wob, nullptr, (float*)d_out, 2);
}
